// NALayer_6270652252945
// MI455X (gfx1250) — compile-verified, matrix operand order checked by reading, not run on hardware
//
#include <hip/hip_runtime.h>
#include <hip/hip_bf16.h>

#define HEADS 4
#define HD    16
#define KS    7
#define KK    (KS*KS)
#define NB    8
#define CH    64
#define HH    64
#define WW    64
#define TOK   (NB*HH*WW)   /* 32768 tokens */
#define EPSV  1e-5f

typedef __attribute__((ext_vector_type(16))) _Float16 v16h;
typedef __attribute__((ext_vector_type(8)))  _Float16 v8h;
typedef __attribute__((ext_vector_type(8)))  float    v8f;

union Frag { v8h h8[2]; v16h h16; };

// ---------------------------------------------------------------------------
// K0: per-sample mean / rsqrt(var) over C*H*W (instance-wide norm stats)
// ---------------------------------------------------------------------------
__global__ void stats_kernel(const float* __restrict__ x, float* __restrict__ stats) {
    int n = blockIdx.x;
    const float* xs = x + (size_t)n * CH * HH * WW;
    float s = 0.f, s2 = 0.f;
    for (int i = threadIdx.x; i < CH * HH * WW; i += blockDim.x) {
        float v = xs[i];
        s += v; s2 += v * v;
    }
    __shared__ float sh1[256], sh2[256];
    sh1[threadIdx.x] = s; sh2[threadIdx.x] = s2;
    __syncthreads();
    for (int off = 128; off > 0; off >>= 1) {
        if ((int)threadIdx.x < off) {
            sh1[threadIdx.x] += sh1[threadIdx.x + off];
            sh2[threadIdx.x] += sh2[threadIdx.x + off];
        }
        __syncthreads();
    }
    if (threadIdx.x == 0) {
        float inv = 1.f / (float)(CH * HH * WW);
        float mu  = sh1[0] * inv;
        float var = sh2[0] * inv - mu * mu;
        stats[2 * n]     = mu;
        stats[2 * n + 1] = rsqrtf(var + EPSV);
    }
}

// ---------------------------------------------------------------------------
// K1: normalize + affine, NCHW -> NHWC; keep fp32 residual x1 and f16 xg
// ---------------------------------------------------------------------------
__global__ void norm_tr_kernel(const float* __restrict__ x, const float* __restrict__ gw,
                               const float* __restrict__ gb, const float* __restrict__ stats,
                               _Float16* __restrict__ xg, float* __restrict__ x1) {
    int idx = blockIdx.x * blockDim.x + threadIdx.x;   // NCHW linear (coalesced read)
    if (idx >= NB * CH * HH * WW) return;
    int w = idx & 63, h = (idx >> 6) & 63, c = (idx >> 12) & 63, n = idx >> 18;
    float v = x[idx];
    int t = (n * HH + h) * WW + w;
    x1[(size_t)t * CH + c] = v;
    float g = (v - stats[2 * n]) * stats[2 * n + 1] * gw[c] + gb[c];
    xg[(size_t)t * CH + c] = (_Float16)g;
}

// ---------------------------------------------------------------------------
// K2: fp32 -> f16 weight conversion (weights kept in native (O,K) row-major)
// ---------------------------------------------------------------------------
__global__ void cvt_kernel(const float* __restrict__ in, _Float16* __restrict__ out, int n) {
    int i = blockIdx.x * blockDim.x + threadIdx.x;
    if (i < n) out[i] = (_Float16)in[i];
}

// ---------------------------------------------------------------------------
// WMMA GEMM: out[M x NDIM] = A[M x KDIM] (f16 row-major) * W^T (+bias, +epi)
// W is (NDIM, KDIM) row-major, so B-fragment columns are contiguous rows of W.
//
// Block = 8 waves; the block async-stages the whole weight panel into LDS
// (GLOBAL_LOAD_ASYNC_TO_LDS_B128, ASYNCcnt) with 16B row padding for bank-
// conflict-free ds_load_b128 B-fragment reads. Each wave owns one 16-row
// M-strip: A fragments preloaded once into registers, then it sweeps all
// N-tiles reading B from LDS -> A read once, W read once per block.
//
// EPI: 0 = store f16 | 1 = +res, store f32 | 2 = exact GELU, store f16
//      3 = +res, store f32 transposed to NCHW
// ---------------------------------------------------------------------------
template <int KDIM, int NDIM, int EPI>
__global__ __launch_bounds__(256) void gemm_wmma_kernel(
    const _Float16* __restrict__ A, const _Float16* __restrict__ W,
    const float* __restrict__ bias, const float* __restrict__ res,
    void* __restrict__ out)
{
    constexpr int NT      = NDIM / 16;
    constexpr int KSTEPS  = KDIM / 32;
    constexpr int LROW    = KDIM + 8;            // +16B pad: odd # of 16B chunks
    constexpr int CH_ROW  = KDIM / 8;            // 16B chunks per weight row
    constexpr int NCHUNK  = NDIM * CH_ROW;       // total 16B chunks to stage

    __shared__ _Float16 wlds[NDIM * LROW];

    // ---- async-stage W (NDIM x KDIM f16) into padded LDS rows --------------
    {
        unsigned ldsBase = (unsigned)(uintptr_t)wlds;
        for (int cid = threadIdx.x; cid < NCHUNK; cid += 256) {
            int row = cid / CH_ROW;
            int c16 = cid - row * CH_ROW;
            unsigned l = ldsBase + (unsigned)(row * LROW + c16 * 8) * 2u;
            const void* g = (const char*)(W + (size_t)row * KDIM + c16 * 8);
            asm volatile("global_load_async_to_lds_b128 %0, %1, off"
                         :: "v"(l), "v"(g) : "memory");
        }
        asm volatile("s_wait_asynccnt 0" ::: "memory");
    }
    __syncthreads();

    int wave = (blockIdx.x * blockDim.x + threadIdx.x) >> 5;
    int lane = threadIdx.x & 31;
    int l15 = lane & 15, half = lane >> 4;
    int m0 = wave * 16;

    // ---- preload all A fragments for this 16-row strip ---------------------
    // lane(l15,half) holds row m0+l15, K = k0+half*8..+8 and k0+16+half*8..+8
    Frag fa[KSTEPS];
    const _Float16* arow = A + (size_t)(m0 + l15) * KDIM + half * 8;
#pragma unroll
    for (int ks = 0; ks < KSTEPS; ++ks) {
        fa[ks].h8[0] = *(const v8h*)(arow + ks * 32);
        fa[ks].h8[1] = *(const v8h*)(arow + ks * 32 + 16);
    }

    int mbase = m0 + half * 8;
#pragma unroll
    for (int nt = 0; nt < NT; ++nt) {
        // B-frag: lane(l15,half) holds col nt*16+l15 (= padded LDS row), same K split
        const _Float16* brow = wlds + (size_t)(nt * 16 + l15) * LROW + half * 8;
        v8f acc = {};
#pragma unroll
        for (int ks = 0; ks < KSTEPS; ++ks) {
            Frag fb;
            fb.h8[0] = *(const v8h*)(brow + ks * 32);
            fb.h8[1] = *(const v8h*)(brow + ks * 32 + 16);
            acc = __builtin_amdgcn_wmma_f32_16x16x32_f16(false, fa[ks].h16, false, fb.h16,
                                                         (short)0, acc, false, false);
        }

        // C/D layout: lane<16 -> col rows m0+0..7; lane>=16 -> rows m0+8..15
        int col  = nt * 16 + l15;
        float bv = bias[col];
#pragma unroll
        for (int r = 0; r < 8; ++r) {
            int m = mbase + r;
            float v = acc[r] + bv;
            if constexpr (EPI == 0) {
                ((_Float16*)out)[(size_t)m * NDIM + col] = (_Float16)v;
            } else if constexpr (EPI == 1) {
                v += res[(size_t)m * NDIM + col];
                ((float*)out)[(size_t)m * NDIM + col] = v;
            } else if constexpr (EPI == 2) {
                v = 0.5f * v * (1.f + erff(v * 0.70710678118f));   // exact GELU
                ((_Float16*)out)[(size_t)m * NDIM + col] = (_Float16)v;
            } else {                                               // EPI == 3
                v += res[(size_t)m * NDIM + col];
                int n = m >> 12, h = (m >> 6) & 63, w = m & 63;
                ((float*)out)[(((size_t)n * CH + col) * HH + h) * WW + w] = v;
            }
        }
    }
}

// ---------------------------------------------------------------------------
// K4: neighborhood attention. One thread per (token, head).
// qkv rows: [q(0..63) | k(64..127) | v(128..191)], per head 16 f16 values.
// ---------------------------------------------------------------------------
__global__ __launch_bounds__(256) void attn_kernel(const _Float16* __restrict__ qkv,
                                                   const float* __restrict__ rpb,
                                                   _Float16* __restrict__ aout)
{
    int tid = blockIdx.x * blockDim.x + threadIdx.x;
    if (tid >= TOK * HEADS) return;
    int head = tid & 3;
    int t    = tid >> 2;
    int n = t >> 12, i = (t >> 6) & 63, j = t & 63;
    int si = i - 3; si = si < 0 ? 0 : si; si = si > HH - KS ? HH - KS : si;
    int sj = j - 3; sj = sj < 0 ? 0 : sj; sj = sj > WW - KS ? WW - KS : sj;
    int dh = i - si + 6, dw = j - sj + 6;   // rel index = dh - a, dw - b

    const _Float16* qp = qkv + (size_t)t * 192 + head * 16;
    float q[HD];
#pragma unroll
    for (int d = 0; d < HD; ++d) q[d] = (float)qp[d] * 0.25f;   // hd^-0.5

    __builtin_prefetch(qkv + (size_t)t * 192 + 128, 0, 0);      // global_prefetch_b8

    const float* rp = rpb + head * 13 * 13;
    float s[KK];
    float mx = -1e30f;
#pragma unroll
    for (int a = 0; a < KS; ++a) {
#pragma unroll
        for (int b = 0; b < KS; ++b) {
            int tk = (n * HH + si + a) * WW + sj + b;
            const v8h* kp = (const v8h*)(qkv + (size_t)tk * 192 + 64 + head * 16);
            v8h k0 = kp[0], k1 = kp[1];
            float dot = 0.f;
#pragma unroll
            for (int d = 0; d < 8; ++d) dot += q[d] * (float)k0[d];
#pragma unroll
            for (int d = 0; d < 8; ++d) dot += q[d + 8] * (float)k1[d];
            dot += rp[(dh - a) * 13 + (dw - b)];
            s[a * KS + b] = dot;
            mx = fmaxf(mx, dot);
        }
    }
    float sum = 0.f;
#pragma unroll
    for (int e = 0; e < KK; ++e) { s[e] = __expf(s[e] - mx); sum += s[e]; }
    float inv = 1.f / sum;

    float o[HD];
#pragma unroll
    for (int d = 0; d < HD; ++d) o[d] = 0.f;
#pragma unroll
    for (int a = 0; a < KS; ++a) {
#pragma unroll
        for (int b = 0; b < KS; ++b) {
            float wgt = s[a * KS + b] * inv;
            int tk = (n * HH + si + a) * WW + sj + b;
            const v8h* vp = (const v8h*)(qkv + (size_t)tk * 192 + 128 + head * 16);
            v8h v0 = vp[0], v1 = vp[1];
#pragma unroll
            for (int d = 0; d < 8; ++d) o[d]     += wgt * (float)v0[d];
#pragma unroll
            for (int d = 0; d < 8; ++d) o[d + 8] += wgt * (float)v1[d];
        }
    }
    _Float16* op = aout + (size_t)t * 64 + head * 16;
#pragma unroll
    for (int d = 0; d < HD; ++d) op[d] = (_Float16)o[d];
}

// ---------------------------------------------------------------------------
// K6: LayerNorm over C=64 per token; one wave32 per token (2 channels/lane)
// ---------------------------------------------------------------------------
__global__ void ln_kernel(const float* __restrict__ y, const float* __restrict__ lw,
                          const float* __restrict__ lb, _Float16* __restrict__ yn)
{
    int wid  = (blockIdx.x * blockDim.x + threadIdx.x) >> 5;
    int lane = threadIdx.x & 31;
    if (wid >= TOK) return;
    const float* yr = y + (size_t)wid * 64;
    float v0 = yr[lane], v1 = yr[lane + 32];
    float s = v0 + v1, s2 = v0 * v0 + v1 * v1;
    for (int m = 16; m > 0; m >>= 1) {
        s  += __shfl_xor(s,  m, 32);
        s2 += __shfl_xor(s2, m, 32);
    }
    float mu  = s * (1.f / 64.f);
    float var = s2 * (1.f / 64.f) - mu * mu;
    float rs  = rsqrtf(var + EPSV);
    _Float16* o = yn + (size_t)wid * 64;
    o[lane]      = (_Float16)((v0 - mu) * rs * lw[lane]      + lb[lane]);
    o[lane + 32] = (_Float16)((v1 - mu) * rs * lw[lane + 32] + lb[lane + 32]);
}

// ---------------------------------------------------------------------------
extern "C" void kernel_launch(void* const* d_in, const int* in_sizes, int n_in,
                              void* d_out, int out_size, void* d_ws, size_t ws_size,
                              hipStream_t stream)
{
    (void)in_sizes; (void)n_in; (void)out_size; (void)ws_size;

    const float* x      = (const float*)d_in[0];
    const float* gn_w   = (const float*)d_in[1];
    const float* gn_b   = (const float*)d_in[2];
    const float* qkv_w  = (const float*)d_in[3];
    const float* qkv_b  = (const float*)d_in[4];
    const float* rpb    = (const float*)d_in[5];
    const float* proj_w = (const float*)d_in[6];
    const float* proj_b = (const float*)d_in[7];
    const float* ln_w   = (const float*)d_in[8];
    const float* ln_b   = (const float*)d_in[9];
    const float* fc1_w  = (const float*)d_in[10];
    const float* fc1_b  = (const float*)d_in[11];
    const float* fc2_w  = (const float*)d_in[12];
    const float* fc2_b  = (const float*)d_in[13];
    float* out = (float*)d_out;

    char* ws = (char*)d_ws;
    size_t off = 0;
    auto alloc = [&](size_t bytes) {
        void* p = ws + off;
        off = (off + bytes + 255) & ~(size_t)255;
        return p;
    };
    float*    stats = (float*)    alloc(64);
    _Float16* xg    = (_Float16*) alloc((size_t)TOK * 64 * 2);
    float*    x1    = (float*)    alloc((size_t)TOK * 64 * 4);
    _Float16* wq    = (_Float16*) alloc((size_t)192 * 64 * 2);
    _Float16* wp    = (_Float16*) alloc((size_t)64 * 64 * 2);
    _Float16* w1    = (_Float16*) alloc((size_t)256 * 64 * 2);
    _Float16* w2    = (_Float16*) alloc((size_t)64 * 256 * 2);
    _Float16* qkvh  = (_Float16*) alloc((size_t)TOK * 192 * 2);
    _Float16* aouth = (_Float16*) alloc((size_t)TOK * 64 * 2);
    float*    yf    = (float*)    alloc((size_t)TOK * 64 * 4);
    _Float16* ynh   = (_Float16*) alloc((size_t)TOK * 64 * 2);
    _Float16* hmid  = (_Float16*) alloc((size_t)TOK * 256 * 2);

    stats_kernel<<<NB, 256, 0, stream>>>(x, stats);
    norm_tr_kernel<<<(NB * CH * HH * WW) / 256, 256, 0, stream>>>(x, gn_w, gn_b, stats, xg, x1);
    cvt_kernel<<<(192 * 64 + 255) / 256, 256, 0, stream>>>(qkv_w, wq, 192 * 64);
    cvt_kernel<<<(64 * 64  + 255) / 256, 256, 0, stream>>>(proj_w, wp, 64 * 64);
    cvt_kernel<<<(256 * 64 + 255) / 256, 256, 0, stream>>>(fc1_w, w1, 256 * 64);
    cvt_kernel<<<(64 * 256 + 255) / 256, 256, 0, stream>>>(fc2_w, w2, 64 * 256);

    // 2048 M-strips, 8 waves per block -> 256 blocks per GEMM
    gemm_wmma_kernel<64, 192, 0><<<256, 256, 0, stream>>>(xg, wq, qkv_b, nullptr, qkvh);
    attn_kernel<<<(TOK * HEADS) / 256, 256, 0, stream>>>(qkvh, rpb, aouth);
    gemm_wmma_kernel<64, 64, 1><<<256, 256, 0, stream>>>(aouth, wp, proj_b, x1, yf);
    ln_kernel<<<(TOK * 32) / 256, 256, 0, stream>>>(yf, ln_w, ln_b, ynh);
    gemm_wmma_kernel<64, 256, 2><<<256, 256, 0, stream>>>(ynh, w1, fc1_b, nullptr, hmid);
    gemm_wmma_kernel<256, 64, 3><<<256, 256, 0, stream>>>(hmid, w2, fc2_b, yf, out);
}
